// CoAttention_71803263254920
// MI455X (gfx1250) — matrix operand order repaired and checked by reading, not compiled, not run on hardware
//
#include <hip/hip_runtime.h>
#include <hip/hip_bf16.h>

// Problem constants (from reference)
#define BB   64
#define CC   1024
#define CK   128
#define NN   256

typedef __attribute__((ext_vector_type(16))) __bf16 v16bf;
typedef __attribute__((ext_vector_type(8)))  __bf16 bf16x8;
typedef __attribute__((ext_vector_type(4)))  __bf16 bf16x4;
typedef __attribute__((ext_vector_type(8)))  float  v8f;

union V16U { v16bf v; bf16x8 h[2]; };

#define LDSK 40   // padded LDS K-stride (elements): 80B rows, 16B aligned

// A-matrix 16x32 bf16 fragment (ISA 7.12.2): lanes 0-15 -> rows M=0..15, K {0..7,16..23};
// lanes 16-31 -> same rows, K {8..15,24..31}.
__device__ __forceinline__ v16bf load_a_frag(const __bf16* base, int row0, int lane) {
  const int half = lane >> 4;
  const __bf16* p = base + (size_t)(row0 + (lane & 15)) * LDSK + half * 8;
  V16U u;
  u.h[0] = *(const bf16x8*)(p);
  u.h[1] = *(const bf16x8*)(p + 16);
  return u.v;
}

// B-matrix 32x16 bf16 fragment: lanes 0-15 -> col N=lane, K {0..7,16..23}; lanes 16-31 -> K {8..15,24..31}.
// LDS tile stored as [outputCol][K] so each lane reads two runs of 8 contiguous bf16 (K base and K base + 16).
__device__ __forceinline__ v16bf load_b_frag(const __bf16* base, int col0, int lane) {
  const __bf16* p = base + (size_t)(col0 + (lane & 15)) * LDSK + (lane >> 4) * 8;
  V16U u;
  u.h[0] = *(const bf16x8*)(p);
  u.h[1] = *(const bf16x8*)(p + 16);
  return u.v;
}

#define WMMA_BF16(a, b, c) \
  __builtin_amdgcn_wmma_f32_16x16x32_bf16(false, (a), false, (b), (short)0, (c), false, false)

// Shared 8-wave 128x128 compute step: 2x4 fragments per wave, 8 WMMAs.
#define COMPUTE_STEP(As, Bs)                                   \
  {                                                            \
    v16bf a0 = load_a_frag((As), wm + 0,  lane);               \
    v16bf a1 = load_a_frag((As), wm + 16, lane);               \
    v16bf b0 = load_b_frag((Bs), wn + 0,  lane);               \
    v16bf b1 = load_b_frag((Bs), wn + 16, lane);               \
    v16bf b2 = load_b_frag((Bs), wn + 32, lane);               \
    v16bf b3 = load_b_frag((Bs), wn + 48, lane);               \
    acc[0][0] = WMMA_BF16(a0, b0, acc[0][0]);                  \
    acc[0][1] = WMMA_BF16(a0, b1, acc[0][1]);                  \
    acc[0][2] = WMMA_BF16(a0, b2, acc[0][2]);                  \
    acc[0][3] = WMMA_BF16(a0, b3, acc[0][3]);                  \
    acc[1][0] = WMMA_BF16(a1, b0, acc[1][0]);                  \
    acc[1][1] = WMMA_BF16(a1, b1, acc[1][1]);                  \
    acc[1][2] = WMMA_BF16(a1, b2, acc[1][2]);                  \
    acc[1][3] = WMMA_BF16(a1, b3, acc[1][3]);                  \
  }

#define ZERO_ACC()                                             \
  _Pragma("unroll")                                            \
  for (int i = 0; i < 2; ++i)                                  \
    _Pragma("unroll")                                          \
    for (int j = 0; j < 4; ++j)                                \
      _Pragma("unroll")                                        \
      for (int r = 0; r < 8; ++r) acc[i][j][r] = 0.0f;

// ---------------------------------------------------------------------------
// Projection: O = W @ X + bias (per batch). W fp32 [M][CC], X fp32 [B][CC][NN].
// fp32 -> bf16 conversion fused into LDS staging; double-buffered pipeline.
//   transposed==0: Out bf16 [B][M][NN]   (value path)
//   transposed==1: Out bf16 [B][NN][M]   (key path, K-contiguous for energy GEMM)
// ---------------------------------------------------------------------------
__global__ __launch_bounds__(256) void proj_kernel(
    const float* __restrict__ W, const float* __restrict__ bias,
    const float* __restrict__ X, __bf16* __restrict__ Out,
    int M, int transposed)
{
  __shared__ __bf16 Asl[2][128 * LDSK];
  __shared__ __bf16 Bsl[2][128 * LDSK];

  const int b    = blockIdx.z;
  const int m0   = blockIdx.x * 128;
  const int n0   = blockIdx.y * 128;
  const int tid  = threadIdx.x;
  const int wave = tid >> 5, lane = tid & 31;
  const int wm   = (wave >> 1) * 32;   // 4 waves along M
  const int wn   = (wave & 1) * 64;    // 2 waves along N

  const float* Xb = X + (size_t)b * CC * NN;

  // Per-thread staging geometry
  const int arow = tid >> 1, ahalf = (tid & 1) * 16;     // A: 128 rows x 32 K
  const int bkb  = (tid & 7) * 4, bnb = (tid >> 3) * 4;  // B: 4x4 micro-blocks

  v8f acc[2][4];
  ZERO_ACC();

  // stage(k0 -> buf): fp32 global loads, cvt, packed LDS stores
  #define PROJ_STAGE(buf, k0)                                                    \
  {                                                                              \
    const float* g = W + (size_t)(m0 + arow) * CC + (k0) + ahalf;                \
    float4 f0 = *(const float4*)(g + 0);                                         \
    float4 f1 = *(const float4*)(g + 4);                                         \
    float4 f2 = *(const float4*)(g + 8);                                         \
    float4 f3 = *(const float4*)(g + 12);                                        \
    bf16x8 p0, p1;                                                               \
    p0[0]=(__bf16)f0.x; p0[1]=(__bf16)f0.y; p0[2]=(__bf16)f0.z; p0[3]=(__bf16)f0.w; \
    p0[4]=(__bf16)f1.x; p0[5]=(__bf16)f1.y; p0[6]=(__bf16)f1.z; p0[7]=(__bf16)f1.w; \
    p1[0]=(__bf16)f2.x; p1[1]=(__bf16)f2.y; p1[2]=(__bf16)f2.z; p1[3]=(__bf16)f2.w; \
    p1[4]=(__bf16)f3.x; p1[5]=(__bf16)f3.y; p1[6]=(__bf16)f3.z; p1[7]=(__bf16)f3.w; \
    __bf16* s = Asl[buf] + (size_t)arow * LDSK + ahalf;                          \
    *(bf16x8*)(s)     = p0;                                                      \
    *(bf16x8*)(s + 8) = p1;                                                      \
    /* B transposed: Bsl[n][k] = X[k0+k][n0+n], 4x4 block per thread */          \
    float fb[4][4];                                                              \
    _Pragma("unroll")                                                            \
    for (int i = 0; i < 4; ++i) {                                                \
      float4 f = *(const float4*)(Xb + (size_t)((k0) + bkb + i) * NN + n0 + bnb);\
      fb[i][0]=f.x; fb[i][1]=f.y; fb[i][2]=f.z; fb[i][3]=f.w;                    \
    }                                                                            \
    _Pragma("unroll")                                                            \
    for (int j = 0; j < 4; ++j) {                                                \
      bf16x4 pk;                                                                 \
      _Pragma("unroll")                                                          \
      for (int i = 0; i < 4; ++i) pk[i] = (__bf16)fb[i][j];                      \
      *(bf16x4*)(Bsl[buf] + (size_t)(bnb + j) * LDSK + bkb) = pk;                \
    }                                                                            \
  }

  const int ITERS = CC / 32;  // 32
  PROJ_STAGE(0, 0);
  for (int it = 0; it < ITERS; ++it) {
    __syncthreads();                      // staging of buf[it&1] complete
    if (it + 1 < ITERS) PROJ_STAGE((it + 1) & 1, (it + 1) * 32);
    COMPUTE_STEP(Asl[it & 1], Bsl[it & 1]);
  }
  #undef PROJ_STAGE

  // Epilogue. C/D layout: VGPR j, lanes 0-15 -> (M=j, N=lane); lanes 16-31 -> (M=j+8, N=lane-16).
  const int colL = lane & 15;
  const int rsel = (lane >> 4) * 8;
  #pragma unroll
  for (int i = 0; i < 2; ++i) {
    #pragma unroll
    for (int j = 0; j < 4; ++j) {
      const int mb = m0 + wm + i * 16 + rsel;
      const int ng = n0 + wn + j * 16 + colL;
      if (!transposed) {
        __bf16* o = Out + (size_t)b * M * NN + ng;
        #pragma unroll
        for (int r = 0; r < 8; ++r)
          o[(size_t)(mb + r) * NN] = (__bf16)(acc[i][j][r] + bias[mb + r]);
      } else {
        bf16x8 pk;
        #pragma unroll
        for (int r = 0; r < 8; ++r) pk[r] = (__bf16)(acc[i][j][r] + bias[mb + r]);
        *(bf16x8*)(Out + (size_t)b * NN * M + (size_t)ng * M + mb) = pk; // 16B packed store
      }
    }
  }
}

// ---------------------------------------------------------------------------
// Energy: E[n][m] = sum_k kxT[n][k] * kyT[m][k]  (K = CK = 128), fp32 out.
// Double-buffered; sources already bf16 K-contiguous.
// ---------------------------------------------------------------------------
__global__ __launch_bounds__(256) void energy_kernel(
    const __bf16* __restrict__ KX, const __bf16* __restrict__ KY,
    float* __restrict__ E)
{
  __shared__ __bf16 Asl[2][128 * LDSK];
  __shared__ __bf16 Bsl[2][128 * LDSK];

  const int b   = blockIdx.z;
  const int r0  = blockIdx.x * 128;
  const int c0  = blockIdx.y * 128;
  const int tid = threadIdx.x;
  const int wave = tid >> 5, lane = tid & 31;
  const int wm = (wave >> 1) * 32, wn = (wave & 1) * 64;

  const __bf16* Ab = KX + (size_t)b * NN * CK;
  const __bf16* Bb = KY + (size_t)b * NN * CK;
  const int row = tid >> 1, half = (tid & 1) * 16;

  v8f acc[2][4];
  ZERO_ACC();

  #define EN_STAGE(buf, k0)                                                       \
  {                                                                               \
    bf16x8 a0 = *(const bf16x8*)(Ab + (size_t)(r0 + row) * CK + (k0) + half);     \
    bf16x8 a1 = *(const bf16x8*)(Ab + (size_t)(r0 + row) * CK + (k0) + half + 8); \
    bf16x8 b0v = *(const bf16x8*)(Bb + (size_t)(c0 + row) * CK + (k0) + half);    \
    bf16x8 b1v = *(const bf16x8*)(Bb + (size_t)(c0 + row) * CK + (k0) + half + 8);\
    *(bf16x8*)(Asl[buf] + (size_t)row * LDSK + half)     = a0;                    \
    *(bf16x8*)(Asl[buf] + (size_t)row * LDSK + half + 8) = a1;                    \
    *(bf16x8*)(Bsl[buf] + (size_t)row * LDSK + half)     = b0v;                   \
    *(bf16x8*)(Bsl[buf] + (size_t)row * LDSK + half + 8) = b1v;                   \
  }

  const int ITERS = CK / 32;  // 4
  EN_STAGE(0, 0);
  for (int it = 0; it < ITERS; ++it) {
    __syncthreads();
    if (it + 1 < ITERS) EN_STAGE((it + 1) & 1, (it + 1) * 32);
    COMPUTE_STEP(Asl[it & 1], Bsl[it & 1]);
  }
  #undef EN_STAGE

  float* Eb = E + (size_t)b * NN * NN;
  const int colL = lane & 15, rsel = (lane >> 4) * 8;
  #pragma unroll
  for (int i = 0; i < 2; ++i)
    #pragma unroll
    for (int j = 0; j < 4; ++j) {
      const int rb = r0 + wm + i * 16 + rsel;
      const int cg = c0 + wn + j * 16 + colL;
      #pragma unroll
      for (int r = 0; r < 8; ++r)
        Eb[(size_t)(rb + r) * NN + cg] = acc[i][j][r];
    }
}

// ---------------------------------------------------------------------------
// Dual softmax from the single E (energy_y == E^T):
//   P1[m][n] = attn_x[m][n] = exp(E[m][n]-rowmax[m]) / rowsum[m]
//   P2[m][n] = attn_y[m][n] = exp(E[n][m]-colmax[m]) / colsum[m]
// ---------------------------------------------------------------------------
__global__ __launch_bounds__(256) void softmax_kernel(
    const float* __restrict__ E, __bf16* __restrict__ P1, __bf16* __restrict__ P2)
{
  __shared__ float rmax[NN], rinv[NN], cmax[NN], cinv[NN];
  const int b = blockIdx.x, t = threadIdx.x;
  const float* Eb = E + (size_t)b * NN * NN;

  float m = -3.0e38f;
  for (int i = 0; i < NN; ++i) m = fmaxf(m, Eb[(size_t)t * NN + i]);
  float s = 0.0f;
  for (int i = 0; i < NN; ++i) s += __expf(Eb[(size_t)t * NN + i] - m);
  rmax[t] = m; rinv[t] = 1.0f / s;

  m = -3.0e38f;
  for (int i = 0; i < NN; ++i) m = fmaxf(m, Eb[(size_t)i * NN + t]);
  s = 0.0f;
  for (int i = 0; i < NN; ++i) s += __expf(Eb[(size_t)i * NN + t] - m);
  cmax[t] = m; cinv[t] = 1.0f / s;
  __syncthreads();

  const float rm = rmax[t], ri = rinv[t], cm = cmax[t], ci = cinv[t];
  __bf16* p1 = P1 + (size_t)b * NN * NN + (size_t)t * NN;
  __bf16* p2 = P2 + (size_t)b * NN * NN + (size_t)t * NN;
  for (int n = 0; n < NN; ++n) {
    p1[n] = (__bf16)(__expf(Eb[(size_t)t * NN + n] - rm) * ri);
    p2[n] = (__bf16)(__expf(Eb[(size_t)n * NN + t] - cm) * ci);
  }
}

// ---------------------------------------------------------------------------
// Output GEMM + epilogue: Out[b][c][m] = gamma * sum_n V[b][c][n]*P[m][n] + X[b][c][m]
// A = V bf16 [CC][NN], B tile = P rows [m][k=n]. K = NN = 256. Double-buffered.
// ---------------------------------------------------------------------------
__global__ __launch_bounds__(256) void outgemm_kernel(
    const __bf16* __restrict__ V, const __bf16* __restrict__ P,
    const float* __restrict__ Xres, const float* __restrict__ gamma,
    float* __restrict__ Out)
{
  __shared__ __bf16 Asl[2][128 * LDSK];
  __shared__ __bf16 Bsl[2][128 * LDSK];

  const int b   = blockIdx.z;
  const int m0  = blockIdx.x * 128;   // channel tile
  const int n0  = blockIdx.y * 128;   // output-column tile
  const int tid = threadIdx.x;
  const int wave = tid >> 5, lane = tid & 31;
  const int wm = (wave >> 1) * 32, wn = (wave & 1) * 64;

  const __bf16* Ab = V + (size_t)b * CC * NN;
  const __bf16* Bb = P + (size_t)b * NN * NN;
  const int row = tid >> 1, half = (tid & 1) * 16;

  v8f acc[2][4];
  ZERO_ACC();

  #define OG_STAGE(buf, k0)                                                       \
  {                                                                               \
    bf16x8 a0 = *(const bf16x8*)(Ab + (size_t)(m0 + row) * NN + (k0) + half);     \
    bf16x8 a1 = *(const bf16x8*)(Ab + (size_t)(m0 + row) * NN + (k0) + half + 8); \
    bf16x8 b0v = *(const bf16x8*)(Bb + (size_t)(n0 + row) * NN + (k0) + half);    \
    bf16x8 b1v = *(const bf16x8*)(Bb + (size_t)(n0 + row) * NN + (k0) + half + 8);\
    *(bf16x8*)(Asl[buf] + (size_t)row * LDSK + half)     = a0;                    \
    *(bf16x8*)(Asl[buf] + (size_t)row * LDSK + half + 8) = a1;                    \
    *(bf16x8*)(Bsl[buf] + (size_t)row * LDSK + half)     = b0v;                   \
    *(bf16x8*)(Bsl[buf] + (size_t)row * LDSK + half + 8) = b1v;                   \
  }

  const int ITERS = NN / 32;  // 8
  OG_STAGE(0, 0);
  for (int it = 0; it < ITERS; ++it) {
    __syncthreads();
    if (it + 1 < ITERS) OG_STAGE((it + 1) & 1, (it + 1) * 32);
    COMPUTE_STEP(Asl[it & 1], Bsl[it & 1]);
  }
  #undef OG_STAGE

  const float g = gamma[0];
  const float* Xb = Xres + (size_t)b * CC * NN;
  float* Ob = Out + (size_t)b * CC * NN;
  const int colL = lane & 15, rsel = (lane >> 4) * 8;
  #pragma unroll
  for (int i = 0; i < 2; ++i)
    #pragma unroll
    for (int j = 0; j < 4; ++j) {
      const int cb = m0 + wm + i * 16 + rsel;
      const int ng = n0 + wn + j * 16 + colL;
      #pragma unroll
      for (int r = 0; r < 8; ++r) {
        const size_t idx = (size_t)(cb + r) * NN + ng;
        Ob[idx] = g * acc[i][j][r] + Xb[idx];
      }
    }
}

// ---------------------------------------------------------------------------
extern "C" void kernel_launch(void* const* d_in, const int* in_sizes, int n_in,
                              void* d_out, int out_size, void* d_ws, size_t ws_size,
                              hipStream_t stream) {
  const float* x      = (const float*)d_in[0];
  const float* y      = (const float*)d_in[1];
  const float* wk1    = (const float*)d_in[2];
  const float* bk1    = (const float*)d_in[3];
  const float* wk2    = (const float*)d_in[4];
  const float* bk2    = (const float*)d_in[5];
  const float* wv1    = (const float*)d_in[6];
  const float* bv1    = (const float*)d_in[7];
  const float* wv2    = (const float*)d_in[8];
  const float* bv2    = (const float*)d_in[9];
  const float* gamma1 = (const float*)d_in[10];
  const float* gamma2 = (const float*)d_in[11];

  // Workspace carve-up (~104 MiB total, L2-resident on MI455X)
  char* ws = (char*)d_ws;
  const size_t szKT = (size_t)BB * NN * CK * sizeof(__bf16);   // 4 MiB
  const size_t szV  = (size_t)BB * CC * NN * sizeof(__bf16);   // 32 MiB
  const size_t szE  = (size_t)BB * NN * NN * sizeof(float);    // 16 MiB
  const size_t szP  = (size_t)BB * NN * NN * sizeof(__bf16);   // 8 MiB
  __bf16* kxT = (__bf16*)(ws);
  __bf16* kyT = (__bf16*)(ws + szKT);
  __bf16* vxb = (__bf16*)(ws + 2 * szKT);
  __bf16* vyb = (__bf16*)(ws + 2 * szKT + szV);
  float*  E   = (float*) (ws + 2 * szKT + 2 * szV);
  __bf16* P1  = (__bf16*)(ws + 2 * szKT + 2 * szV + szE);
  __bf16* P2  = (__bf16*)(ws + 2 * szKT + 2 * szV + szE + szP);

  float* out_x = (float*)d_out;
  float* out_y = out_x + (size_t)BB * CC * NN;

  dim3 blk(256);
  // Projections (keys stored transposed [N][CK] for K-contiguous energy GEMM)
  proj_kernel<<<dim3(1, 2, BB), blk, 0, stream>>>(wk1, bk1, x, kxT, CK, 1);
  proj_kernel<<<dim3(1, 2, BB), blk, 0, stream>>>(wk2, bk2, y, kyT, CK, 1);
  proj_kernel<<<dim3(8, 2, BB), blk, 0, stream>>>(wv1, bv1, x, vxb, CC, 0);
  proj_kernel<<<dim3(8, 2, BB), blk, 0, stream>>>(wv2, bv2, y, vyb, CC, 0);
  // Single energy GEMM (energy_y is its transpose)
  energy_kernel<<<dim3(2, 2, BB), blk, 0, stream>>>(kxT, kyT, E);
  // Dual softmax producing both attention operands in GEMM-ready layout
  softmax_kernel<<<dim3(BB), blk, 0, stream>>>(E, P1, P2);
  // Output GEMMs with fused gamma*o + residual epilogue
  outgemm_kernel<<<dim3(8, 2, BB), blk, 0, stream>>>(vxb, P1, x, gamma1, out_x);
  outgemm_kernel<<<dim3(8, 2, BB), blk, 0, stream>>>(vyb, P2, y, gamma2, out_y);
}
